// Global_31971736551587
// MI455X (gfx1250) — matrix operand order repaired and checked by reading, not compiled, not run on hardware
//
#include <hip/hip_runtime.h>
#include <math.h>

typedef __attribute__((ext_vector_type(16))) __bf16 v16bf;
typedef __attribute__((ext_vector_type(8)))  float  v8f;

#define DIM 256
#define HEADS 4
#define CHAN 64
#define NEG_SLOPE 0.2f

__device__ __forceinline__ __bf16 f2bf(float f) {
    union { float f; unsigned u; } v; v.f = f;
    unsigned short top = (unsigned short)((v.u + 0x7FFFu + ((v.u >> 16) & 1u)) >> 16);
    union { unsigned short s; __bf16 b; } o; o.s = top;
    return o.b;
}

__device__ __forceinline__ void atomicMaxF(float* addr, float val) {
    if (val >= 0.f) atomicMax((int*)addr, __float_as_int(val));
    else            atomicMin((unsigned int*)addr, __float_as_uint(val));
}

__device__ __forceinline__ float gelu_exact(float x) {
    return 0.5f * x * (1.f + erff(x * 0.70710678118654752f));
}

// ---------------------------------------------------------------------------
// Y[S,256] = X[S,256] @ W[256,256], bf16 WMMA, f32 accumulate.
// Block: 256 threads = 8 waves; block tile 64 rows x 256 cols.
// Wave w: row-tile = w>>1 (16 rows), columns (w&1)*128 .. +127 (8 col-tiles).
// ---------------------------------------------------------------------------
__global__ void __launch_bounds__(256) gemm_bf16_256(
        const float* __restrict__ X, const float* __restrict__ W,
        float* __restrict__ Y, int S) {
    __shared__ __align__(32) __bf16 Xs[64][32];    //  4 KB, row-major [m][k]
    __shared__ __align__(32) __bf16 Ws[256][32];   // 16 KB, transposed  [n][k]

    const int t       = threadIdx.x;
    const int rowBase = blockIdx.x * 64;
    const int wave    = t >> 5;
    const int lane    = t & 31;
    const int hl      = lane >> 4;      // lane half: 0 | 1
    const int mrow    = lane & 15;
    const int rowTile = wave >> 1;      // 0..3
    const int colBase = (wave & 1) * 128;

    v8f acc[8] = {};

    for (int k0 = 0; k0 < DIM; k0 += 32) {
        // stage X tile: thread t -> row t>>2, k (t&3)*8 .. +7
        {
            const int r  = t >> 2;
            const int kk = (t & 3) * 8;
            const int gr = rowBase + r;
            #pragma unroll
            for (int j = 0; j < 8; ++j) {
                float v = (gr < S) ? X[gr * DIM + k0 + kk + j] : 0.f;
                Xs[r][kk + j] = f2bf(v);
            }
        }
        // stage W tile transposed: thread t == output column n
        {
            #pragma unroll
            for (int k = 0; k < 32; ++k)
                Ws[t][k] = f2bf(W[(k0 + k) * DIM + t]);
        }
        __syncthreads();

        // A fragment (16x32 bf16): lanes 0-15: K = {0..7,16..23}; lanes 16-31: +8
        v16bf a;
        {
            const __bf16* base = &Xs[rowTile * 16 + mrow][0];
            #pragma unroll
            for (int j = 0; j < 8; ++j) {
                a[j]     = base[hl * 8 + j];
                a[j + 8] = base[16 + hl * 8 + j];
            }
        }
        // B fragments: lanes 0-15: K = {0..7,16..23}; lanes 16-31: +8, N = mrow
        #pragma unroll
        for (int ct = 0; ct < 8; ++ct) {
            v16bf b;
            const __bf16* bb = &Ws[colBase + ct * 16 + mrow][0];
            #pragma unroll
            for (int j = 0; j < 8; ++j) { b[j] = bb[hl * 8 + j]; b[j + 8] = bb[16 + hl * 8 + j]; }
            acc[ct] = __builtin_amdgcn_wmma_f32_16x16x32_bf16(
                false, a, false, b, (short)0, acc[ct], false, false);
        }
        __syncthreads();
    }

    // C/D layout: VGPR v -> M = v + 8*half, N = lane&15
    #pragma unroll
    for (int ct = 0; ct < 8; ++ct) {
        const int col = colBase + ct * 16 + mrow;
        #pragma unroll
        for (int v = 0; v < 8; ++v) {
            const int row = rowBase + rowTile * 16 + hl * 8 + v;
            if (row < S) Y[row * DIM + col] = acc[ct][v];
        }
    }
}

// ---------------------------------------------------------------------------
// init stats: emax = -inf, esum = 0  [n_dst*HEADS]
// ---------------------------------------------------------------------------
__global__ void init_stats(float* __restrict__ emax, float* __restrict__ esum, int n4) {
    const int idx = blockIdx.x * blockDim.x + threadIdx.x;
    if (idx < n4) { emax[idx] = -__builtin_inff(); esum[idx] = 0.f; }
}

__global__ void zero_buf(float* __restrict__ p, int n) {
    const int idx = blockIdx.x * blockDim.x + threadIdx.x;
    if (idx < n) p[idx] = 0.f;
}

// ---------------------------------------------------------------------------
// edge score: wave per edge.  e[h] = sum_c att[h,c]*lrelu(xl[s,h,c]+xr[d,h,c])
// ---------------------------------------------------------------------------
__global__ void __launch_bounds__(256) edge_score(
        const float* __restrict__ xl, const float* __restrict__ xr,
        const int* __restrict__ src, const int* __restrict__ dst,
        const float* __restrict__ att, float* __restrict__ eval,
        float* __restrict__ emax, int E) {
    const int e    = blockIdx.x * 8 + (threadIdx.x >> 5);
    const int lane = threadIdx.x & 31;
    if (e >= E) return;
    const int s = src[e], d = dst[e];
    const float2* pxl = (const float2*)(xl + (long long)s * DIM);
    const float2* pxr = (const float2*)(xr + (long long)d * DIM);
    const float2* pat = (const float2*)att;
    #pragma unroll
    for (int h = 0; h < HEADS; ++h) {
        float2 a = pxl[h * 32 + lane];
        float2 b = pxr[h * 32 + lane];
        float2 w = pat[h * 32 + lane];
        float z0 = a.x + b.x; z0 = (z0 > 0.f) ? z0 : NEG_SLOPE * z0;
        float z1 = a.y + b.y; z1 = (z1 > 0.f) ? z1 : NEG_SLOPE * z1;
        float p = z0 * w.x + z1 * w.y;
        #pragma unroll
        for (int off = 16; off; off >>= 1) p += __shfl_xor(p, off, 32);
        if (lane == h) {
            eval[e * HEADS + h] = p;
            atomicMaxF(&emax[d * HEADS + h], p);
        }
    }
}

// ---------------------------------------------------------------------------
// exp(e - max) + segment sum.  thread per (edge, head)
// ---------------------------------------------------------------------------
__global__ void edge_exp(const int* __restrict__ dst, float* __restrict__ eval,
                         const float* __restrict__ emax, float* __restrict__ esum,
                         int E) {
    const int idx = blockIdx.x * blockDim.x + threadIdx.x;
    if (idx >= E * HEADS) return;
    const int d = dst[idx >> 2];
    const int h = idx & 3;
    const float ex = expf(eval[idx] - emax[d * HEADS + h]);
    eval[idx] = ex;
    atomicAdd(&esum[d * HEADS + h], ex);
}

// ---------------------------------------------------------------------------
// aggregate: wave per edge; agg[d] += xl[s] * alpha   (float atomics, L2)
// ---------------------------------------------------------------------------
__global__ void __launch_bounds__(256) edge_aggregate(
        const float* __restrict__ xl, const int* __restrict__ src,
        const int* __restrict__ dst, const float* __restrict__ eval,
        const float* __restrict__ esum, float* __restrict__ agg, int E) {
    const int e    = blockIdx.x * 8 + (threadIdx.x >> 5);
    const int lane = threadIdx.x & 31;
    if (e >= E) return;
    const int s = src[e], d = dst[e];
    const float2* pxl = (const float2*)(xl + (long long)s * DIM);
    float* base = agg + (long long)d * DIM;
    #pragma unroll
    for (int h = 0; h < HEADS; ++h) {
        const float alpha = eval[e * HEADS + h] / (esum[d * HEADS + h] + 1e-16f);
        float2 a = pxl[h * 32 + lane];
        atomicAdd(&base[h * CHAN + lane * 2 + 0], a.x * alpha);
        atomicAdd(&base[h * CHAN + lane * 2 + 1], a.y * alpha);
    }
}

// ---------------------------------------------------------------------------
// epilogue: dest += gelu(agg + bias)
// ---------------------------------------------------------------------------
__global__ void epilogue(const float* __restrict__ agg, const float* __restrict__ bias,
                         float* __restrict__ dest, int n_dst) {
    const int idx = blockIdx.x * blockDim.x + threadIdx.x;
    if (idx >= n_dst * DIM) return;
    const float v = agg[idx] + bias[idx & (DIM - 1)];
    dest[idx] += gelu_exact(v);
}

// ---------------------------------------------------------------------------
static void run_conv2(const float* x_src, int S_src, const float* x_dst, int S_dst,
                      const int* src, const int* dst, int E,
                      const float* Wl, const float* Wr, const float* att,
                      const float* bias, float* dest,
                      float* xl, float* xr, float* eval,
                      float* emax, float* esum, hipStream_t stream) {
    const int n_dst = S_dst;
    gemm_bf16_256<<<(S_src + 63) / 64, 256, 0, stream>>>(x_src, Wl, xl, S_src);
    gemm_bf16_256<<<(S_dst + 63) / 64, 256, 0, stream>>>(x_dst, Wr, xr, S_dst);
    init_stats<<<(n_dst * HEADS + 255) / 256, 256, 0, stream>>>(emax, esum, n_dst * HEADS);
    edge_score<<<(E + 7) / 8, 256, 0, stream>>>(xl, xr, src, dst, att, eval, emax, E);
    // xr is dead after edge_score -> reuse as aggregation buffer
    zero_buf<<<(n_dst * DIM + 255) / 256, 256, 0, stream>>>(xr, n_dst * DIM);
    edge_exp<<<(E * HEADS + 255) / 256, 256, 0, stream>>>(dst, eval, emax, esum, E);
    edge_aggregate<<<(E + 7) / 8, 256, 0, stream>>>(xl, src, dst, eval, esum, xr, E);
    epilogue<<<(n_dst * DIM + 255) / 256, 256, 0, stream>>>(xr, bias, dest, n_dst);
}

extern "C" void kernel_launch(void* const* d_in, const int* in_sizes, int n_in,
                              void* d_out, int out_size, void* d_ws, size_t ws_size,
                              hipStream_t stream) {
    const float* node_emb = (const float*)d_in[0];
    const float* hedge_emb= (const float*)d_in[1];
    const float* Wl0 = (const float*)d_in[2];
    const float* Wr0 = (const float*)d_in[3];
    const float* att0= (const float*)d_in[4];
    const float* b0  = (const float*)d_in[5];
    const float* Wl1 = (const float*)d_in[6];
    const float* Wr1 = (const float*)d_in[7];
    const float* att1= (const float*)d_in[8];
    const float* b1  = (const float*)d_in[9];
    const int* fwd_src = (const int*)d_in[10];
    const int* fwd_dst = (const int*)d_in[11];
    const int* bwd_src = (const int*)d_in[12];
    const int* bwd_dst = (const int*)d_in[13];

    const int N  = in_sizes[0] / DIM;
    const int M  = in_sizes[1] / DIM;
    const int Ef = in_sizes[10];
    const int Eb = in_sizes[12];
    const int maxSM = (N > M) ? N : M;
    const int maxE  = (Ef > Eb) ? Ef : Eb;
    const int L = in_sizes[2] / (DIM * DIM);

    float* ws   = (float*)d_ws;
    float* hbuf = ws;                              // M*256
    float* xl   = hbuf + (size_t)M * DIM;          // maxSM*256
    float* xr   = xl   + (size_t)maxSM * DIM;      // maxSM*256 (also agg)
    float* eval = xr   + (size_t)maxSM * DIM;      // maxE*4
    float* emax = eval + (size_t)maxE * HEADS;     // maxSM*4
    float* esum = emax + (size_t)maxSM * HEADS;    // maxSM*4
    float* nbuf = (float*)d_out;                   // n lives in d_out

    (void)n_in; (void)ws_size; (void)out_size;

    (void)hipMemcpyAsync(nbuf, node_emb, (size_t)N * DIM * sizeof(float),
                         hipMemcpyDeviceToDevice, stream);
    (void)hipMemcpyAsync(hbuf, hedge_emb, (size_t)M * DIM * sizeof(float),
                         hipMemcpyDeviceToDevice, stream);

    for (int l = 0; l < L; ++l) {
        // forward: nodes -> hyperedges, h = h + gelu(conv)
        run_conv2(nbuf, N, hbuf, M, fwd_src, fwd_dst, Ef,
                  Wl0 + (size_t)l * DIM * DIM, Wr0 + (size_t)l * DIM * DIM,
                  att0 + (size_t)l * DIM, b0 + (size_t)l * DIM,
                  hbuf, xl, xr, eval, emax, esum, stream);
        // backward: hyperedges -> nodes, n = n + gelu(conv)
        run_conv2(hbuf, M, nbuf, N, bwd_src, bwd_dst, Eb,
                  Wl1 + (size_t)l * DIM * DIM, Wr1 + (size_t)l * DIM * DIM,
                  att1 + (size_t)l * DIM, b1 + (size_t)l * DIM,
                  nbuf, xl, xr, eval, emax, esum, stream);
    }
}
